// PaiConv_10050223472786
// MI455X (gfx1250) — matrix operand order repaired and checked by reading, not compiled, not run on hardware
//
#include <hip/hip_runtime.h>
#include <hip/hip_bf16.h>

// ---------- types ----------
typedef __attribute__((ext_vector_type(16))) __bf16 v16bf;
typedef __attribute__((ext_vector_type(8)))  __bf16 v8bf;
typedef __attribute__((ext_vector_type(2)))  __bf16 v2bf;
typedef __attribute__((ext_vector_type(8)))  float  v8f;
typedef __attribute__((ext_vector_type(2)))  float  v2f;
typedef __attribute__((ext_vector_type(8)))  unsigned int v8u;

// ---------- helpers ----------
__device__ __forceinline__ unsigned short f32_to_bf16_us(float f) {
    return __builtin_bit_cast(unsigned short, (__bf16)f);   // native v_cvt on gfx1250
}
// True packed convert: one v_cvt_pk_bf16_f32 for a pair
__device__ __forceinline__ unsigned int pack_bf16(float lo, float hi) {
    v2f f = {lo, hi};
    return __builtin_bit_cast(unsigned int, __builtin_convertvector(f, v2bf));
}

// Problem constants (B=16, C=64, N=4096, K=20, KS=20, OC=64)
#define NPTS      65536        // B*N
#define KNB       20
#define KSD       20
#define OCH       64
#define CKS       1280         // C*KS
#define KPAD      32           // stage-1 K padded for WMMA
#define PTS_PER_BLK 16

// ---------- kernel 0: prep (weights -> bf16, K-permuted f'=s*64+c; zero BN sums) ----------
__global__ __launch_bounds__(256) void prep_kernel(const float* __restrict__ conv_w,
                                                   unsigned short* __restrict__ wbf,
                                                   float* __restrict__ sums) {
    int i = blockIdx.x * 256 + threadIdx.x;          // 81920 weights
    if (i < OCH * CKS) {
        int oc = i / CKS, fp = i % CKS;
        int s = fp >> 6, c = fp & 63;                // f' = s*64 + c
        wbf[i] = f32_to_bf16_us(conv_w[oc * CKS + c * KSD + s]);
    }
    if (i < 128) sums[i] = 0.0f;
}

// ---------- kernel 1: fused gather + einsum + conv (both stages on WMMA) ----------
__global__ __launch_bounds__(128) void paiconv_main(const float* __restrict__ feature,
                                                    const float* __restrict__ permatrix,
                                                    const float* __restrict__ conv_b,
                                                    const int*   __restrict__ neigh,
                                                    const unsigned short* __restrict__ wbf,
                                                    float* __restrict__ out,
                                                    float* __restrict__ sums) {
    // perm_T[pt][s][k] : k padded to 32 (pad MUST be zeroed: WMMA propagates NaN)
    __shared__ __align__(16) unsigned short perm_T[PTS_PER_BLK * KSD * KPAD];   // 20 KB
    // feats2T[pt][s][c] : stage-1 output, f' = s*64 + c
    __shared__ __align__(16) unsigned short feats2T[PTS_PER_BLK * CKS];         // 40 KB
    __shared__ int nidx_lds[PTS_PER_BLK * KNB];                                 // 1.25 KB

    const int tile = blockIdx.x;                 // 16 points per block
    const int tid  = threadIdx.x;
    const int lane = tid & 31;
    const int wave = tid >> 5;                   // 4 waves
    const int kg   = lane >> 4;                  // K-group (0/1) per WMMA layout
    const int ln   = lane & 15;

    // ---- pass 0: zero perm_T (protects the k>=20 pad from NaN garbage) ----
    {
        unsigned int* pz = (unsigned int*)perm_T;
        for (int e = tid; e < (PTS_PER_BLK * KSD * KPAD) / 2; e += 128) pz[e] = 0u;
    }
    __syncthreads();

    // ---- pass 1: stage permatrix transposed (->bf16) + neighbor indices ----
    {
        const float* psrc = permatrix + tile * (PTS_PER_BLK * KNB * KSD);
        for (int e = tid; e < PTS_PER_BLK * KNB * KSD; e += 128) {
            int pt = e / (KNB * KSD), rem = e % (KNB * KSD);
            int k = rem / KSD, s = rem % KSD;
            perm_T[pt * (KSD * KPAD) + s * KPAD + k] = f32_to_bf16_us(psrc[e]);
        }
        for (int e = tid; e < PTS_PER_BLK * KNB; e += 128)
            nidx_lds[e] = neigh[tile * (PTS_PER_BLK * KNB) + e];
    }
    __syncthreads();

    // ================= Stage 1: per-point feats(64x20) @ perm(20x20) =================
    // A-element pair validity (pair jj covers j=2jj,2jj+1):
    //   jj=0..3 : k = kg*8 + j            -> always valid (k<16)
    //   jj=4,5  : k = 16 + kg*8 + (j-8)   -> valid only for kg==0 (k=16..19 vs 24..27)
    //   jj=6,7  : k = 20..23 / 28..31     -> never valid (compile-time zero, no load)
    for (int i = 0; i < 4; ++i) {
        const int pt = wave * 4 + i;

        // B tiles: element j -> K = kg*8 + j (j<8), 16 + kg*8 + (j-8) (j>=8), N(s) = ln / 16+ln.
        // Tile-1 rows s>=20 clamp to row 0: finite garbage lands only in discarded D cols.
        const unsigned short* pbase = perm_T + pt * (KSD * KPAD) + kg * 8;
        v8bf b0l = *reinterpret_cast<const v8bf*>(pbase + ln * KPAD);
        v8bf b0h = *reinterpret_cast<const v8bf*>(pbase + ln * KPAD + 16);
        v16bf b0 = __builtin_shufflevector(b0l, b0h, 0, 1, 2, 3, 4, 5, 6, 7, 8, 9, 10, 11, 12, 13, 14, 15);
        const int s1row = (ln < KSD - 16) ? (16 + ln) : 0;
        v8bf b1l = *reinterpret_cast<const v8bf*>(pbase + s1row * KPAD);
        v8bf b1h = *reinterpret_cast<const v8bf*>(pbase + s1row * KPAD + 16);
        v16bf b1 = __builtin_shufflevector(b1l, b1h, 0, 1, 2, 3, 4, 5, 6, 7, 8, 9, 10, 11, 12, 13, 14, 15);

        // Gather base addresses for j=0..11 (j>=12 never used). For kg=1, j=8..11 the
        // index is clamped to 0 (safe in-bounds address); result masked after pack.
        int abase[12];
        #pragma unroll
        for (int j = 0; j < 12; ++j) {
            int v = j >> 1, h = j & 1;
            int k = ((v >= 4) ? 16 : 0) + kg * 8 + (v & 3) * 2 + h;
            int idx = (k < KNB) ? nidx_lds[pt * KNB + k] : 0;
            abase[j] = ((idx >> 12) << 18) + (idx & 4095);      // b*C*N + n
        }

        unsigned short* f2base  = feats2T + pt * CKS + ln * 64;           // s = ln
        unsigned short* f2base1 = feats2T + pt * CKS + (16 + ln) * 64;    // s = 16+ln (ln<4)

        #pragma unroll
        for (int mt = 0; mt < 4; ++mt) {
            const int m    = mt * 16 + ln;                      // permuted channel row
            const int in_c = ((m & 3) << 4) | (m >> 2);         // group-interleave inverse
            const int coff = in_c << 12;

            v8u au;
            #pragma unroll
            for (int jj = 0; jj < 4; ++jj) {                    // always-valid pairs
                float f0 = feature[abase[2 * jj]     + coff];
                float f1 = feature[abase[2 * jj + 1] + coff];
                au[jj] = pack_bf16(f0, f1);
            }
            #pragma unroll
            for (int jj = 4; jj < 6; ++jj) {                    // valid only for kg==0
                float f0 = feature[abase[2 * jj]     + coff];
                float f1 = feature[abase[2 * jj + 1] + coff];
                unsigned int p = pack_bf16(f0, f1);
                au[jj] = (kg == 0) ? p : 0u;                    // one v_cndmask
            }
            au[6] = 0u;                                         // never valid
            au[7] = 0u;
            v16bf a = __builtin_bit_cast(v16bf, au);

            v8f c0 = {}, c1 = {};
            c0 = __builtin_amdgcn_wmma_f32_16x16x32_bf16(false, a, false, b0, (short)0, c0, false, false);
            c1 = __builtin_amdgcn_wmma_f32_16x16x32_bf16(false, a, false, b1, (short)0, c1, false, false);

            // D rows r -> consecutive c = mt*16 + kg*8 + r at fixed s: one b128 store.
            const int cbase = mt * 16 + kg * 8;
            uint4 q0;
            q0.x = pack_bf16(c0[0], c0[1]);
            q0.y = pack_bf16(c0[2], c0[3]);
            q0.z = pack_bf16(c0[4], c0[5]);
            q0.w = pack_bf16(c0[6], c0[7]);
            *reinterpret_cast<uint4*>(f2base + cbase) = q0;

            if (ln < KSD - 16) {                                // s = 16+ln < 20 (4 lanes)
                uint4 q1;
                q1.x = pack_bf16(c1[0], c1[1]);
                q1.y = pack_bf16(c1[2], c1[3]);
                q1.z = pack_bf16(c1[4], c1[5]);
                q1.w = pack_bf16(c1[6], c1[7]);
                *reinterpret_cast<uint4*>(f2base1 + cbase) = q1;
            }
        }
    }
    __syncthreads();

    // ================= Stage 2: points(16) x weights^T -> OC(64) =================
    // M = 16 points, N = 16 OC per wave, K = 1280 exact over f' = s*64+c.
    // A fragment = two aligned ds_load_b128; B = one 32-byte contiguous global load.
    const int oc = wave * 16 + ln;
    const unsigned short* arow = feats2T + ln * CKS;
    const unsigned short* wrow = wbf + oc * CKS + kg * 16;

    v8f acc = {};
    #pragma unroll 4
    for (int step = 0; step < CKS / 32; ++step) {
        __builtin_prefetch(wrow + (step + 2) * 32, 0, 1);        // global_prefetch_b8

        v16bf b = *reinterpret_cast<const v16bf*>(wrow + step * 32);

        v8bf alo = *reinterpret_cast<const v8bf*>(arow + step * 32 + kg * 16);
        v8bf ahi = *reinterpret_cast<const v8bf*>(arow + step * 32 + kg * 16 + 8);
        v16bf a  = __builtin_shufflevector(alo, ahi, 0, 1, 2, 3, 4, 5, 6, 7,
                                                     8, 9, 10, 11, 12, 13, 14, 15);

        acc = __builtin_amdgcn_wmma_f32_16x16x32_bf16(false, a, false, b, (short)0, acc, false, false);
    }

    // bias + write pre-BN output in [B, OC, N] layout + BN partial sums
    const float bias = conv_b[oc];
    float s1 = 0.0f, s2 = 0.0f;
    #pragma unroll
    for (int r = 0; r < 8; ++r) {
        float val = acc[r] + bias;
        int pg = tile * PTS_PER_BLK + r + kg * 8;                // global point id
        int b_ = pg >> 12, n = pg & 4095;
        out[((b_ << 6) + oc) * 4096 + n] = val;
        s1 += val;
        s2 += val * val;
    }
    s1 += __shfl_xor(s1, 16, 32);
    s2 += __shfl_xor(s2, 16, 32);
    if (kg == 0) {
        atomicAdd(&sums[oc], s1);
        atomicAdd(&sums[64 + oc], s2);
    }
}

// ---------- kernel 2: fold BN stats into scale/shift ----------
__global__ __launch_bounds__(64) void bn_stats(const float* __restrict__ sums,
                                               const float* __restrict__ gamma,
                                               const float* __restrict__ beta,
                                               float* __restrict__ sc) {
    int oc = threadIdx.x;
    if (oc < OCH) {
        const float inv_n = 1.0f / (float)NPTS;
        float m   = sums[oc] * inv_n;
        float var = sums[64 + oc] * inv_n - m * m;
        float s   = gamma[oc] * rsqrtf(var + 1e-5f);
        sc[oc]      = s;
        sc[64 + oc] = beta[oc] - m * s;
    }
}

// ---------- kernel 3: in-place BN apply on [B, OC, N], float4 vectorized ----------
__global__ __launch_bounds__(256) void bn_apply(float4* __restrict__ out,
                                                const float* __restrict__ sc) {
    int i  = blockIdx.x * 256 + threadIdx.x;     // 1,048,576 float4s
    int oc = (i >> 10) & 63;                     // 1024 float4 per (b,oc) row
    float s = sc[oc], t = sc[64 + oc];
    float4 v = out[i];
    v.x = v.x * s + t;
    v.y = v.y * s + t;
    v.z = v.z * s + t;
    v.w = v.w * s + t;
    out[i] = v;
}

// ---------- host launcher ----------
extern "C" void kernel_launch(void* const* d_in, const int* in_sizes, int n_in,
                              void* d_out, int out_size, void* d_ws, size_t ws_size,
                              hipStream_t stream) {
    const float* feature   = (const float*)d_in[0];   // [16,64,4096]
    const float* permatrix = (const float*)d_in[1];   // [65536,20,20]
    const float* conv_w    = (const float*)d_in[2];   // [64,1280]
    const float* conv_b    = (const float*)d_in[3];   // [64]
    const float* gamma     = (const float*)d_in[4];   // [64]
    const float* beta      = (const float*)d_in[5];   // [64]
    const int*   neigh     = (const int*)d_in[6];     // [65536,20]
    float* out = (float*)d_out;                       // [16,64,4096]

    // ws layout: [0,163840): weights bf16 (K-permuted); [163840,164352): sums; [164352,164864): scale/shift
    unsigned short* wbf  = (unsigned short*)d_ws;
    float*          sums = (float*)((char*)d_ws + 163840);
    float*          sc   = (float*)((char*)d_ws + 164352);

    prep_kernel<<<320, 256, 0, stream>>>(conv_w, wbf, sums);
    paiconv_main<<<NPTS / PTS_PER_BLK, 128, 0, stream>>>(feature, permatrix, conv_b, neigh,
                                                         wbf, out, sums);
    bn_stats<<<1, 64, 0, stream>>>(sums, gamma, beta, sc);
    bn_apply<<<(NPTS * OCH) / 4 / 256, 256, 0, stream>>>((float4*)out, sc);
}
